// Model_64209761075918
// MI455X (gfx1250) — matrix operand order repaired and checked by reading, not compiled, not run on hardware
//
#include <hip/hip_runtime.h>
#include <stdint.h>

typedef __attribute__((ext_vector_type(16))) _Float16 v16h;
typedef __attribute__((ext_vector_type(8)))  float    v8f;
typedef __attribute__((ext_vector_type(4)))  uint32_t u32x4;

#define BATCHN   8
#define SEQ_LEN  1024
#define ENC_IN   7
#define PRED_LEN 96
#define D_MODEL  512
#define N_HEADS  8
#define D_HEAD   64
#define D_FF     2048
#define E_LAYERS 2
#define PATCH    16
#define STRIDEP  8
#define N_PATCH  128
#define NF       (D_MODEL * N_PATCH)   // 65536
#define BC       (BATCHN * ENC_IN)     // 56
#define ROWS     (BC * N_PATCH)        // 7168
#define U_TOP    25
#define ZBH      (BC * N_HEADS)        // 448

// ---------------------------------------------------------------- mean / std
__global__ void k_meanstd(const float* __restrict__ x, float* mean, float* stdv) {
    int bc = blockIdx.x; int b = bc / ENC_IN, c = bc % ENC_IN;
    __shared__ float s1[256], s2[256];
    float a = 0.f, q = 0.f;
    for (int t = threadIdx.x; t < SEQ_LEN; t += 256) {
        float v = x[(size_t)(b * SEQ_LEN + t) * ENC_IN + c];
        a += v; q += v * v;
    }
    s1[threadIdx.x] = a; s2[threadIdx.x] = q; __syncthreads();
    for (int s = 128; s > 0; s >>= 1) {
        if (threadIdx.x < s) { s1[threadIdx.x] += s1[threadIdx.x + s]; s2[threadIdx.x] += s2[threadIdx.x + s]; }
        __syncthreads();
    }
    if (threadIdx.x == 0) {
        float m = s1[0] / SEQ_LEN;
        float var = s2[0] / SEQ_LEN - m * m;
        mean[bc] = m; stdv[bc] = sqrtf(var + 1e-5f);
    }
}

// ---------------------------------------------- normalize + patch embed + PE
__global__ void k_patch_embed(const float* __restrict__ x, const float* __restrict__ pW,
                              const float* __restrict__ mean, const float* __restrict__ stdv,
                              float* __restrict__ tok, _Float16* __restrict__ tok16) {
    int bc = blockIdx.x;
    int idx = blockIdx.y * blockDim.x + threadIdx.x;      // 0..65535
    int p = idx >> 9, d = idx & 511;
    int b = bc / ENC_IN, c = bc % ENC_IN;
    float m = mean[bc], sd = stdv[bc];
    float acc = 0.f;
#pragma unroll
    for (int j = 0; j < PATCH; j++) {
        int t = p * STRIDEP + j; if (t > SEQ_LEN - 1) t = SEQ_LEN - 1;   // edge pad
        float xv = (x[(size_t)(b * SEQ_LEN + t) * ENC_IN + c] - m) / sd;
        acc += xv * pW[d * PATCH + j];
    }
    int i2 = d >> 1;
    float w = expf(-(float)(2 * i2) * (9.210340371976184f / (float)D_MODEL)); // ln(1e4)
    float ang = (float)p * w;
    acc += (d & 1) ? cosf(ang) : sinf(ang);
    size_t o = (size_t)(bc * N_PATCH + p) * D_MODEL + d;
    tok[o] = acc; tok16[o] = (_Float16)acc;
}

// ------------------------------------------------------------- f32 -> f16
__global__ void k_cvt(const float* __restrict__ a, _Float16* __restrict__ o, long long n) {
    long long i = (long long)blockIdx.x * blockDim.x + threadIdx.x;
    if (i < n) o[i] = (_Float16)a[i];
}

// ------------------------------------------------------- generic WMMA GEMM
// Out[m][n] = scale * sum_k A[m][k]*B[n][k] + bias[n]  (optionally GELU)
// One wave = one 32x32 C tile (2x2 register blocking of 16x16 WMMA subtiles).
// Per 32-wide K step: 8x global_load_b128 feed 4 independent
// v_wmma_f32_16x16x32_f16 accumulator chains (2x operand reuse).
union HV { v16h v; u32x4 q[2]; uint32_t u[8]; };

__global__ void __launch_bounds__(32) k_gemm(
    const _Float16* __restrict__ A, const _Float16* __restrict__ B,
    const float* __restrict__ bias, float* __restrict__ outF, _Float16* __restrict__ outH,
    int M, int N, int K, int lda, int ldb, int ldo,
    long long sA1, long long sA2, long long sB1, long long sB2,
    long long sO1, long long sO2, int bdiv, float scale, int act) {
    int z = blockIdx.z; int i1 = z / bdiv, i2 = z % bdiv;
    const _Float16* Ab = A + (long long)i1 * sA1 + (long long)i2 * sA2;
    const _Float16* Bb = B + (long long)i1 * sB1 + (long long)i2 * sB2;
    long long oOff = (long long)i1 * sO1 + (long long)i2 * sO2;
    int n0 = blockIdx.x * 32, m0 = blockIdx.y * 32;
    int lane = threadIdx.x & 31;
    int hi = lane >> 4;          // lane group (0: lanes 0-15, 1: lanes 16-31)
    int lr = lane & 15;
    int aRow0 = m0 + lr;
    int aRow1 = m0 + 16 + lr;
    // clamp (always-legal address); rows >= M feed lanes whose outputs are masked
    const _Float16* aP0 = Ab + (size_t)(aRow0 < M ? aRow0 : 0) * lda;
    const _Float16* aP1 = Ab + (size_t)(aRow1 < M ? aRow1 : 0) * lda;
    const _Float16* bP0 = Bb + (size_t)(n0 + lr) * ldb;
    const _Float16* bP1 = Bb + (size_t)(n0 + 16 + lr) * ldb;
    int kA = hi * 8;             // A-layout: upper lane group k+8
    int kB = hi * 8;             // B-layout: upper lane group k+8
    v8f acc00 = {}, acc01 = {}, acc10 = {}, acc11 = {};
    for (int kk = 0; kk < K; kk += 32) {
        HV a0, a1, b0, b1;
        const _Float16* pa0 = aP0 + kk + kA;
        const _Float16* pa1 = aP1 + kk + kA;
        const _Float16* pb0 = bP0 + kk + kB;
        const _Float16* pb1 = bP1 + kk + kB;
        a0.q[0] = *(const u32x4*)(pa0);       a0.q[1] = *(const u32x4*)(pa0 + 16);
        a1.q[0] = *(const u32x4*)(pa1);       a1.q[1] = *(const u32x4*)(pa1 + 16);
        b0.q[0] = *(const u32x4*)(pb0);       b0.q[1] = *(const u32x4*)(pb0 + 16);
        b1.q[0] = *(const u32x4*)(pb1);       b1.q[1] = *(const u32x4*)(pb1 + 16);
        acc00 = __builtin_amdgcn_wmma_f32_16x16x32_f16(false, a0.v, false, b0.v,
                                                       (short)0, acc00, false, false);
        acc01 = __builtin_amdgcn_wmma_f32_16x16x32_f16(false, a0.v, false, b1.v,
                                                       (short)0, acc01, false, false);
        acc10 = __builtin_amdgcn_wmma_f32_16x16x32_f16(false, a1.v, false, b0.v,
                                                       (short)0, acc10, false, false);
        acc11 = __builtin_amdgcn_wmma_f32_16x16x32_f16(false, a1.v, false, b1.v,
                                                       (short)0, acc11, false, false);
    }
    int nA = n0 + lr, nB = n0 + 16 + lr;
    float bvA = bias ? bias[nA] : 0.0f;
    float bvB = bias ? bias[nB] : 0.0f;
#pragma unroll
    for (int r = 0; r < 8; r++) {
        int mA = m0 + r + hi * 8;
        int mB = m0 + 16 + r + hi * 8;
        float v00 = acc00[r] * scale + bvA;
        float v01 = acc01[r] * scale + bvB;
        float v10 = acc10[r] * scale + bvA;
        float v11 = acc11[r] * scale + bvB;
        if (act == 1) {
            v00 = 0.5f * v00 * (1.0f + erff(v00 * 0.70710678118654752f));
            v01 = 0.5f * v01 * (1.0f + erff(v01 * 0.70710678118654752f));
            v10 = 0.5f * v10 * (1.0f + erff(v10 * 0.70710678118654752f));
            v11 = 0.5f * v11 * (1.0f + erff(v11 * 0.70710678118654752f));
        }
        if (mA < M) {
            long long o0 = oOff + (long long)mA * ldo + nA;
            long long o1 = oOff + (long long)mA * ldo + nB;
            if (outF) { outF[o0] = v00; outF[o1] = v01; }
            if (outH) { outH[o0] = (_Float16)v00; outH[o1] = (_Float16)v01; }
        }
        if (mB < M) {
            long long o0 = oOff + (long long)mB * ldo + nA;
            long long o1 = oOff + (long long)mB * ldo + nB;
            if (outF) { outF[o0] = v10; outF[o1] = v11; }
            if (outH) { outH[o0] = (_Float16)v10; outH[o1] = (_Float16)v11; }
        }
    }
}

// --------------------------- sparsity measure M = rowmax - rowmean, top-u
__global__ void k_mtopk(const float* __restrict__ S, int* __restrict__ topk) {
    int z = blockIdx.x;
    __shared__ float Mv[N_PATCH];
    int l = threadIdx.x;
    const float* row = S + ((size_t)z * N_PATCH + l) * N_PATCH;
    float mx = -1e30f, sm = 0.f;
    for (int j = 0; j < N_PATCH; j++) { float v = row[j]; mx = fmaxf(mx, v); sm += v; }
    Mv[l] = mx - sm / (float)N_PATCH;
    __syncthreads();
    if (l == 0) {
        for (int i = 0; i < U_TOP; i++) {
            float best = -1e30f; int bi = 0;
            for (int j = 0; j < N_PATCH; j++) if (Mv[j] > best) { best = Mv[j]; bi = j; }
            topk[z * U_TOP + i] = bi; Mv[bi] = -1e31f;
        }
    }
}

// ------------------------------------------------------------ mean of V rows
__global__ void k_vmean(const float* __restrict__ v, float* __restrict__ vm) {
    int z = blockIdx.x; int bc = z / N_HEADS, h = z % N_HEADS; int d = threadIdx.x;
    float s = 0.f;
    for (int n = 0; n < N_PATCH; n++)
        s += v[(size_t)(bc * N_PATCH + n) * D_MODEL + h * D_HEAD + d];
    vm[z * D_HEAD + d] = s / (float)N_PATCH;
}

__global__ void k_ctx_fill(const float* __restrict__ vm, float* __restrict__ ctx) {
    size_t i = (size_t)blockIdx.x * blockDim.x + threadIdx.x;
    if (i >= (size_t)ROWS * D_MODEL) return;
    int col = (int)(i % D_MODEL); size_t rn = i / D_MODEL; int bc = (int)(rn / N_PATCH);
    int h = col / D_HEAD, d = col % D_HEAD;
    ctx[i] = vm[(bc * N_HEADS + h) * D_HEAD + d];
}

// -------------------------------- softmax over selected rows + attn @ V scatter
__global__ void k_softmax_upd(const float* __restrict__ S, const float* __restrict__ v,
                              const int* __restrict__ topk, float* __restrict__ ctx) {
    int z = blockIdx.x, i = blockIdx.y;
    int bc = z / N_HEADS, h = z % N_HEADS;
    int l = topk[z * U_TOP + i];
    __shared__ float a[N_PATCH], red[N_PATCH];
    int t = threadIdx.x;
    float vv = S[((size_t)z * N_PATCH + l) * N_PATCH + t];
    a[t] = vv; red[t] = vv; __syncthreads();
    for (int s = 64; s > 0; s >>= 1) { if (t < s) red[t] = fmaxf(red[t], red[t + s]); __syncthreads(); }
    float mx = red[0]; __syncthreads();
    float e = __expf(a[t] - mx);
    a[t] = e; red[t] = e; __syncthreads();
    for (int s = 64; s > 0; s >>= 1) { if (t < s) red[t] += red[t + s]; __syncthreads(); }
    float inv = 1.0f / red[0]; __syncthreads();
    a[t] *= inv; __syncthreads();
    if (t < D_HEAD) {
        float s = 0.f;
        for (int j = 0; j < N_PATCH; j++)
            s += a[j] * v[(size_t)(bc * N_PATCH + j) * D_MODEL + h * D_HEAD + t];
        ctx[(size_t)(bc * N_PATCH + l) * D_MODEL + h * D_HEAD + t] = s;
    }
}

// ------------------------------------------ (x [+ r]) -> LayerNorm -> f32/f16
__global__ void k_addln(const float* __restrict__ x, const float* __restrict__ r,
                        const float* __restrict__ g, const float* __restrict__ b,
                        float* __restrict__ outF, _Float16* __restrict__ outH) {
    int row = blockIdx.x; int t = threadIdx.x;
    __shared__ float s1[128], s2[128];
    const float* xr = x + (size_t)row * D_MODEL;
    const float* rr = r ? r + (size_t)row * D_MODEL : nullptr;
    float loc[4]; float a = 0.f, q = 0.f;
#pragma unroll
    for (int j = 0; j < 4; j++) {
        int d = t + 128 * j;
        float v = xr[d] + (rr ? rr[d] : 0.0f);
        loc[j] = v; a += v; q += v * v;
    }
    s1[t] = a; s2[t] = q; __syncthreads();
    for (int s = 64; s > 0; s >>= 1) { if (t < s) { s1[t] += s1[t + s]; s2[t] += s2[t + s]; } __syncthreads(); }
    float m = s1[0] / D_MODEL;
    float var = s2[0] / D_MODEL - m * m;
    float inv = rsqrtf(var + 1e-5f);
#pragma unroll
    for (int j = 0; j < 4; j++) {
        int d = t + 128 * j;
        float v = (loc[j] - m) * inv * g[d] + b[d];
        if (outF) outF[(size_t)row * D_MODEL + d] = v;
        if (outH) outH[(size_t)row * D_MODEL + d] = (_Float16)v;
    }
}

// --------------------------- tok (bc,p,d) -> feat16 (bc, d*128+p) for head GEMM
__global__ void k_feat_cvt(const float* __restrict__ tok, _Float16* __restrict__ feat) {
    size_t i = (size_t)blockIdx.x * blockDim.x + threadIdx.x;
    if (i >= (size_t)ROWS * D_MODEL) return;
    int d = (int)(i % D_MODEL); size_t rn = i / D_MODEL;
    int p = (int)(rn % N_PATCH); int bc = (int)(rn / N_PATCH);
    feat[(size_t)bc * NF + (size_t)d * N_PATCH + p] = (_Float16)tok[i];
}

// ------------------------------------------------- de-normalize + transpose out
__global__ void k_head_final(const float* __restrict__ dec, const float* __restrict__ mean,
                             const float* __restrict__ stdv, float* __restrict__ out) {
    int i = blockIdx.x * blockDim.x + threadIdx.x;
    if (i >= BATCHN * PRED_LEN * ENC_IN) return;
    int c = i % ENC_IN; int t = (i / ENC_IN) % PRED_LEN; int b = i / (ENC_IN * PRED_LEN);
    int bc = b * ENC_IN + c;
    out[i] = dec[(size_t)bc * PRED_LEN + t] * stdv[bc] + mean[bc];
}

// ======================================================================
extern "C" void kernel_launch(void* const* d_in, const int* in_sizes, int n_in,
                              void* d_out, int out_size, void* d_ws, size_t ws_size,
                              hipStream_t stream) {
    (void)in_sizes; (void)n_in; (void)out_size; (void)ws_size;
    const float* x_enc  = (const float*)d_in[0];
    const float* patchW = (const float*)d_in[4];
    const float* Wq = (const float*)d_in[5];  const float* bq = (const float*)d_in[6];
    const float* Wk = (const float*)d_in[7];  const float* bk = (const float*)d_in[8];
    const float* Wv = (const float*)d_in[9];  const float* bv = (const float*)d_in[10];
    const float* Wo = (const float*)d_in[11]; const float* bo = (const float*)d_in[12];
    const float* c1W = (const float*)d_in[13]; const float* c1b = (const float*)d_in[14];
    const float* c2W = (const float*)d_in[15]; const float* c2b = (const float*)d_in[16];
    const float* ln1g = (const float*)d_in[17]; const float* ln1b = (const float*)d_in[18];
    const float* ln2g = (const float*)d_in[19]; const float* ln2b = (const float*)d_in[20];
    const float* encg = (const float*)d_in[21]; const float* encb = (const float*)d_in[22];
    const float* headW = (const float*)d_in[23]; const float* headb = (const float*)d_in[24];
    float* out = (float*)d_out;

    char* ws = (char*)d_ws; size_t cur = 0;
    auto carve = [&](size_t bytes) -> char* {
        char* p = ws + cur; cur = (cur + bytes + 255) & ~(size_t)255; return p;
    };
    float*    tok    = (float*)carve((size_t)ROWS * D_MODEL * 4);
    float*    bufQ   = (float*)carve((size_t)ROWS * D_MODEL * 4);   // q, later x1
    float*    bufK   = (float*)carve((size_t)ROWS * D_MODEL * 4);   // k, later ctx
    float*    bufV   = (float*)carve((size_t)ROWS * D_MODEL * 4);
    float*    scores = (float*)carve((size_t)ZBH * N_PATCH * N_PATCH * 4); // also attn_out/y
    _Float16* act16  = (_Float16*)carve((size_t)ROWS * D_MODEL * 2); // tok16/x1-16/ctx16/feat16
    _Float16* ff16   = (_Float16*)carve((size_t)ROWS * D_FF * 2);
    _Float16* q16    = (_Float16*)carve((size_t)ROWS * D_MODEL * 2);
    _Float16* k16    = (_Float16*)carve((size_t)ROWS * D_MODEL * 2);
    _Float16* w16    = (_Float16*)carve((size_t)PRED_LEN * NF * 2);  // largest weight (headW)
    float*    meanB  = (float*)carve(BC * 4);
    float*    stdB   = (float*)carve(BC * 4);
    float*    vmean  = (float*)carve((size_t)ZBH * D_HEAD * 4);
    int*      topk   = (int*)carve((size_t)ZBH * U_TOP * 4);
    float*    decT   = (float*)carve((size_t)BC * PRED_LEN * 4);

    auto cvt = [&](const float* s, _Float16* d, long long n) {
        k_cvt<<<dim3((unsigned)((n + 255) / 256)), 256, 0, stream>>>(s, d, n);
    };
    auto gemm = [&](const _Float16* A, const _Float16* B, const float* bias,
                    float* oF, _Float16* oH, int M, int N, int K,
                    int lda, int ldb, int ldo,
                    long long sA1, long long sA2, long long sB1, long long sB2,
                    long long sO1, long long sO2, int nb, int bdiv,
                    float scale, int act) {
        dim3 g((N + 31) / 32, (M + 31) / 32, nb);
        k_gemm<<<g, 32, 0, stream>>>(A, B, bias, oF, oH, M, N, K, lda, ldb, ldo,
                                     sA1, sA2, sB1, sB2, sO1, sO2, bdiv, scale, act);
    };

    k_meanstd<<<BC, 256, 0, stream>>>(x_enc, meanB, stdB);
    k_patch_embed<<<dim3(BC, (N_PATCH * D_MODEL) / 256), 256, 0, stream>>>(
        x_enc, patchW, meanB, stdB, tok, act16);

    const int TOTE = ROWS * D_MODEL;
    for (int l = 0; l < E_LAYERS; l++) {
        // ---- QKV projections (WMMA), emit f16 copies of Q and K for scores
        cvt(Wq + (size_t)l * D_MODEL * D_MODEL, w16, (long long)D_MODEL * D_MODEL);
        gemm(act16, w16, bq + l * D_MODEL, bufQ, q16, ROWS, D_MODEL, D_MODEL,
             D_MODEL, D_MODEL, D_MODEL, 0, 0, 0, 0, 0, 0, 1, 1, 1.0f, 0);
        cvt(Wk + (size_t)l * D_MODEL * D_MODEL, w16, (long long)D_MODEL * D_MODEL);
        gemm(act16, w16, bk + l * D_MODEL, bufK, k16, ROWS, D_MODEL, D_MODEL,
             D_MODEL, D_MODEL, D_MODEL, 0, 0, 0, 0, 0, 0, 1, 1, 1.0f, 0);
        cvt(Wv + (size_t)l * D_MODEL * D_MODEL, w16, (long long)D_MODEL * D_MODEL);
        gemm(act16, w16, bv + l * D_MODEL, bufV, nullptr, ROWS, D_MODEL, D_MODEL,
             D_MODEL, D_MODEL, D_MODEL, 0, 0, 0, 0, 0, 0, 1, 1, 1.0f, 0);
        // ---- full scores S = Q K^T / sqrt(Dh), batched over (bc, h)
        gemm(q16, k16, nullptr, scores, nullptr, N_PATCH, N_PATCH, D_HEAD,
             D_MODEL, D_MODEL, N_PATCH,
             (long long)N_PATCH * D_MODEL, D_HEAD,
             (long long)N_PATCH * D_MODEL, D_HEAD,
             (long long)N_HEADS * N_PATCH * N_PATCH, (long long)N_PATCH * N_PATCH,
             ZBH, N_HEADS, 0.125f, 0);
        // ---- ProbSparse: measure + top-u, ctx = mean(V) with top rows replaced
        k_mtopk<<<ZBH, N_PATCH, 0, stream>>>(scores, topk);
        k_vmean<<<ZBH, D_HEAD, 0, stream>>>(bufV, vmean);
        k_ctx_fill<<<TOTE / 256, 256, 0, stream>>>(vmean, bufK);
        k_softmax_upd<<<dim3(ZBH, U_TOP), N_PATCH, 0, stream>>>(scores, bufV, topk, bufK);
        // ---- out projection + residual + LN1
        cvt(bufK, act16, (long long)TOTE);   // ctx -> f16
        cvt(Wo + (size_t)l * D_MODEL * D_MODEL, w16, (long long)D_MODEL * D_MODEL);
        gemm(act16, w16, bo + l * D_MODEL, scores, nullptr, ROWS, D_MODEL, D_MODEL,
             D_MODEL, D_MODEL, D_MODEL, 0, 0, 0, 0, 0, 0, 1, 1, 1.0f, 0);
        k_addln<<<ROWS, 128, 0, stream>>>(tok, scores, ln1g + l * D_MODEL, ln1b + l * D_MODEL,
                                          bufQ, act16);
        // ---- FFN: GELU(x1 c1^T + c1b) c2^T + c2b, residual + LN2
        cvt(c1W + (size_t)l * D_FF * D_MODEL, w16, (long long)D_FF * D_MODEL);
        gemm(act16, w16, c1b + l * D_FF, nullptr, ff16, ROWS, D_FF, D_MODEL,
             D_MODEL, D_MODEL, D_FF, 0, 0, 0, 0, 0, 0, 1, 1, 1.0f, 1);
        cvt(c2W + (size_t)l * D_MODEL * D_FF, w16, (long long)D_MODEL * D_FF);
        gemm(ff16, w16, c2b + l * D_MODEL, scores, nullptr, ROWS, D_MODEL, D_FF,
             D_FF, D_FF, D_MODEL, 0, 0, 0, 0, 0, 0, 1, 1, 1.0f, 0);
        k_addln<<<ROWS, 128, 0, stream>>>(bufQ, scores, ln2g + l * D_MODEL, ln2b + l * D_MODEL,
                                          tok, act16);
    }
    // ---- final encoder LN, head GEMM over NF=65536, de-normalize
    k_addln<<<ROWS, 128, 0, stream>>>(tok, nullptr, encg, encb, tok, nullptr);
    k_feat_cvt<<<TOTE / 256, 256, 0, stream>>>(tok, act16);
    cvt(headW, w16, (long long)PRED_LEN * NF);
    gemm(act16, w16, headb, decT, nullptr, BC, PRED_LEN, NF,
         NF, NF, PRED_LEN, 0, 0, 0, 0, 0, 0, 1, 1, 1.0f, 0);
    k_head_final<<<(BATCHN * PRED_LEN * ENC_IN + 255) / 256, 256, 0, stream>>>(
        decT, meanB, stdB, out);
}
